// InterfaceGNN_4252017623370
// MI455X (gfx1250) — matrix operand order repaired and checked by reading, not compiled, not run on hardware
//
#include <hip/hip_runtime.h>
#include <cmath>

// Problem constants (match reference)
#define N_NODES 50000
#define N_EDGES 1600000
#define D_INP   64
#define D_EDGE  16
#define HID     128
#define NLAYERS 6

typedef __attribute__((ext_vector_type(16))) _Float16 v16h;
typedef __attribute__((ext_vector_type(8)))  float    v8f;

__device__ __forceinline__ v8f wmma_f16(v16h a, v16h b, v8f c) {
  // D = A(16x32 f16) * B(32x16 f16) + C(16x16 f32)
  return __builtin_amdgcn_wmma_f32_16x16x32_f16(
      /*neg_a=*/false, a, /*neg_b=*/false, b,
      /*c_mod=*/(short)0, c, /*reuse_a=*/false, /*reuse_b=*/false);
}

// ---------------------------------------------------------------------------
// Kernel 0: column sums of W_e2 / W_as / W_ad  (the algebraic reduction:
//  (z @ W.T).sum(1) == z . colsum(W) + sum(b))
// grid = 6 layers, block = 128 threads (one per k)
// ---------------------------------------------------------------------------
__global__ void prep_kernel(const float* __restrict__ W_e2, const float* __restrict__ b_e2,
                            const float* __restrict__ W_as, const float* __restrict__ b_as,
                            const float* __restrict__ W_ad, const float* __restrict__ b_ad,
                            float* v_e2, float* c_e2,
                            float* v_as, float* c_as,
                            float* v_ad, float* c_ad) {
  int l = blockIdx.x;
  int k = threadIdx.x;
  const float* We2 = W_e2 + (size_t)l * HID * HID;
  const float* Was = W_as + (size_t)l * HID * HID;
  const float* Wad = W_ad + (size_t)l * HID * HID;
  float s2 = 0.f, ss = 0.f, sd = 0.f;
  for (int j = 0; j < HID; ++j) {
    s2 += We2[j * HID + k];
    ss += Was[j * HID + k];
    sd += Wad[j * HID + k];
  }
  v_e2[l * HID + k] = s2;
  v_as[l * HID + k] = ss;
  v_ad[l * HID + k] = sd;
  if (k == 0) {
    float a = 0.f, b = 0.f, c = 0.f;
    for (int j = 0; j < HID; ++j) {
      a += b_e2[l * HID + j];
      b += b_as[l * HID + j];
      c += b_ad[l * HID + j];
    }
    c_e2[l] = a; c_as[l] = b; c_ad[l] = c;
  }
}

// ---------------------------------------------------------------------------
// Kernel 1: per-edge attention bias for ALL 6 layers in one pass over edge_attr.
//   sb[l][e] = relu(tanh(ea[e]) @ W_e1[l].T + b_e1[l]) . colsum(W_e2[l]) + sum(b_e2[l])
// One wave handles a 16-edge tile. All six W_e1 matrices are staged in LDS as
// f16 *zero-padded to K=32*, so every lane's B load is unconditional (no EXEC
// divergence) -- elements 8..15 of each lane read the zero-pad region (K=16..31).
// ---------------------------------------------------------------------------
__global__ void edge_bias_kernel(const float* __restrict__ edge_attr,
                                 const float* __restrict__ W_e1,
                                 const float* __restrict__ b_e1,
                                 const float* __restrict__ v_e2,
                                 const float* __restrict__ c_e2,
                                 float* __restrict__ sb) {
  __shared__ _Float16 sW1[NLAYERS * HID * 32];  // 48 KB, K zero-padded 16->32
  __shared__ float    sB1[NLAYERS * HID];
  __shared__ float    sV2[NLAYERS * HID];
  __shared__ float    sC2[NLAYERS];

  for (int idx = threadIdx.x; idx < NLAYERS * HID * 32; idx += blockDim.x) {
    int k  = idx & 31;
    int rn = idx >> 5;  // l*128 + n
    sW1[idx] = (k < D_EDGE) ? (_Float16)W_e1[rn * D_EDGE + k] : (_Float16)0.f;
  }
  for (int idx = threadIdx.x; idx < NLAYERS * HID; idx += blockDim.x) {
    sB1[idx] = b_e1[idx];
    sV2[idx] = v_e2[idx];
  }
  if (threadIdx.x < NLAYERS) sC2[threadIdx.x] = c_e2[threadIdx.x];
  __syncthreads();

  int wave = (int)(((size_t)blockIdx.x * blockDim.x + threadIdx.x) >> 5);
  int lane = threadIdx.x & 31;
  int e0 = wave * 16;
  if (e0 >= N_EDGES) return;
  int  lrow = lane & 15;
  bool hi   = lane >= 16;

  // A operand: row = edge within tile, K = feature (0..15 real, 16..31 zero pad)
  // layout: element j -> K = (j<8 ? j : j+8) + (hi ? 8 : 0)
  int e  = e0 + lrow;
  int kb = hi ? 8 : 0;
  v16h a;
#pragma unroll
  for (int j = 0; j < 8; ++j)
    a[j] = (_Float16)tanhf(edge_attr[(size_t)e * D_EDGE + kb + j]);
#pragma unroll
  for (int j = 8; j < 16; ++j) a[j] = (_Float16)0.f;   // K=16..31 pad

  for (int l = 0; l < NLAYERS; ++l) {
    float part[8];
#pragma unroll
    for (int r = 0; r < 8; ++r) part[r] = 0.f;

#pragma unroll
    for (int nt = 0; nt < 8; ++nt) {
      int n = nt * 16 + lrow;   // hidden unit = output column
      // B operand: lane = column n, element j -> K = (j<8 ? j : j+8) + (hi ? 8 : 0)
      const _Float16* wp = &sW1[((l * HID) + n) * 32 + (hi ? 8 : 0)];
      v16h b;
#pragma unroll
      for (int j = 0; j < 16; ++j) b[j] = wp[j < 8 ? j : j + 8];
      float bias = sB1[l * HID + n];
      v8f acc;
#pragma unroll
      for (int r = 0; r < 8; ++r) acc[r] = bias;       // fold b_e1 into C
      acc = wmma_f16(a, b, acc);
      float vs = sV2[l * HID + n];
#pragma unroll
      for (int r = 0; r < 8; ++r) part[r] += fmaxf(acc[r], 0.f) * vs;
    }
    // reduce over the 16 columns (lanes 0-15 carry rows 0-7, lanes 16-31 rows 8-15)
#pragma unroll
    for (int r = 0; r < 8; ++r) {
      float p = part[r];
      p += __shfl_xor(p, 1, 16);
      p += __shfl_xor(p, 2, 16);
      p += __shfl_xor(p, 4, 16);
      p += __shfl_xor(p, 8, 16);
      part[r] = p;
    }
    if (lrow == 0) {
      float cadd = sC2[l];
      int mbase = e0 + (hi ? 8 : 0);
#pragma unroll
      for (int r = 0; r < 8; ++r)
        sb[(size_t)l * N_EDGES + mbase + r] = part[r] + cadd;
    }
  }
}

// ---------------------------------------------------------------------------
// Kernel 2: generic node GEMM  D[M,128] = act(A)[M,K] @ W[128,K]^T + bias
// W is staged once per block in LDS as f16 (weight-stationary: all 8 waves
// share it; B-operand feed becomes ds_load_b128). A relu/add applied on load,
// output optionally relu'd. One wave computes a 16x128 output tile.
// ---------------------------------------------------------------------------
template <int KDIM>
__global__ void gemm_kernel(const float* __restrict__ A,
                            const float* __restrict__ add,   // nullable
                            const float* __restrict__ W,     // [128, KDIM]
                            const float* __restrict__ bias,  // [128]
                            float* __restrict__ D,
                            int relu_a, int relu_out) {
  __shared__ _Float16 sW[HID * KDIM];   // 16 KB (K=64) or 32 KB (K=128)
  __shared__ float    sBias[HID];
  for (int idx = threadIdx.x; idx < HID * KDIM; idx += blockDim.x)
    sW[idx] = (_Float16)W[idx];
  if (threadIdx.x < HID) sBias[threadIdx.x] = bias[threadIdx.x];
  __syncthreads();

  constexpr int KCH = KDIM / 32;
  int wave = (int)(((size_t)blockIdx.x * blockDim.x + threadIdx.x) >> 5);
  int lane = threadIdx.x & 31;
  int row0 = wave * 16;
  if (row0 >= N_NODES) return;   // after the staging barrier: safe
  int  lrow = lane & 15;
  bool hi   = lane >= 16;
  int  m    = row0 + lrow;

  v16h a[KCH];
#pragma unroll
  for (int kc = 0; kc < KCH; ++kc) {
    int kb = kc * 32 + (hi ? 8 : 0);
#pragma unroll
    for (int j = 0; j < 16; ++j) {
      int k = kb + (j < 8 ? j : j + 8);
      float v = A[(size_t)m * KDIM + k];
      if (relu_a) v = fmaxf(v, 0.f);
      if (add) v += add[(size_t)m * KDIM + k];
      a[kc][j] = (_Float16)v;
    }
  }

#pragma unroll
  for (int nt = 0; nt < 8; ++nt) {
    int n = nt * 16 + lrow;
    float bv = sBias[n];
    v8f c;
#pragma unroll
    for (int r = 0; r < 8; ++r) c[r] = bv;
#pragma unroll
    for (int kc = 0; kc < KCH; ++kc) {
      const _Float16* wp = &sW[n * KDIM + kc * 32 + (hi ? 8 : 0)];
      v16h b;
#pragma unroll
      for (int j = 0; j < 16; ++j) b[j] = wp[j < 8 ? j : j + 8];
      c = wmma_f16(a[kc], b, c);
    }
    int mo = row0 + (hi ? 8 : 0);
#pragma unroll
    for (int r = 0; r < 8; ++r) {
      float v = c[r];
      if (relu_out) v = fmaxf(v, 0.f);
      D[(size_t)(mo + r) * HID + n] = v;
    }
  }
}

// ---------------------------------------------------------------------------
// Kernel 3: per-node scalars s_src / s_dst (matvec against colsums). Wave/row.
// ---------------------------------------------------------------------------
__global__ void scal_kernel(const float* __restrict__ h,
                            const float* __restrict__ vas, const float* __restrict__ cas,
                            const float* __restrict__ vad, const float* __restrict__ cad,
                            int layer, float* __restrict__ ssrc, float* __restrict__ sdst) {
  int row  = (int)(((size_t)blockIdx.x * blockDim.x + threadIdx.x) >> 5);
  int lane = threadIdx.x & 31;
  if (row >= N_NODES) return;
  float4 hv = *(const float4*)(h + (size_t)row * HID + lane * 4);
  float4 va = *(const float4*)(vas + layer * HID + lane * 4);
  float4 vd = *(const float4*)(vad + layer * HID + lane * 4);
  float h0 = fmaxf(hv.x, 0.f), h1 = fmaxf(hv.y, 0.f);
  float h2 = fmaxf(hv.z, 0.f), h3 = fmaxf(hv.w, 0.f);
  float ps = h0 * va.x + h1 * va.y + h2 * va.z + h3 * va.w;
  float pd = h0 * vd.x + h1 * vd.y + h2 * vd.z + h3 * vd.w;
#pragma unroll
  for (int msk = 1; msk < 32; msk <<= 1) {
    ps += __shfl_xor(ps, msk, 32);
    pd += __shfl_xor(pd, msk, 32);
  }
  if (lane == 0) {
    ssrc[row] = ps + cas[layer];
    sdst[row] = pd + cad[layer];
  }
}

// ---------------------------------------------------------------------------
// Kernel 4: per-edge gather * sigmoid-gate * scatter-add (segment_sum).
// One wave per edge; float4 gather; global_atomic_add_f32 into agg (L2-resident).
// ---------------------------------------------------------------------------
__global__ void edge_kernel(const int* __restrict__ ei, const float* __restrict__ ew,
                            const float* __restrict__ sb,
                            const float* __restrict__ ssrc, const float* __restrict__ sdst,
                            const float* __restrict__ msg, float* __restrict__ agg,
                            int layer) {
  size_t e   = ((size_t)blockIdx.x * blockDim.x + threadIdx.x) >> 5;
  int   lane = threadIdx.x & 31;
  if (e >= N_EDGES) return;
  int src = ei[e];
  int dst = ei[(size_t)N_EDGES + e];
  float t = ssrc[src] + sdst[dst] + sb[(size_t)layer * N_EDGES + e];
  float att = 1.f / (1.f + __expf(-t));
  float w = ew[e];
  w = fminf(fmaxf(w, 0.f), 1.f);
  float scale = att * w;
  float4 m4 = *(const float4*)(msg + (size_t)src * HID + lane * 4);
  float* ag = agg + (size_t)dst * HID + lane * 4;
  unsafeAtomicAdd(ag + 0, m4.x * scale);
  unsafeAtomicAdd(ag + 1, m4.y * scale);
  unsafeAtomicAdd(ag + 2, m4.z * scale);
  unsafeAtomicAdd(ag + 3, m4.w * scale);
}

// ---------------------------------------------------------------------------
// Utility: zero fill
// ---------------------------------------------------------------------------
__global__ void zero_kernel(float* __restrict__ p, size_t n) {
  size_t i = (size_t)blockIdx.x * blockDim.x + threadIdx.x;
  size_t stride = (size_t)gridDim.x * blockDim.x;
  for (; i < n; i += stride) p[i] = 0.f;
}

// ---------------------------------------------------------------------------
// Kernel 5: column sums of h (for the mean) — LDS-staged, then global atomics.
// ---------------------------------------------------------------------------
__global__ void colsum_kernel(const float* __restrict__ h, float* __restrict__ acc) {
  __shared__ float sacc[HID];
  int t = threadIdx.x;  // 256
  if (t < HID) sacc[t] = 0.f;
  __syncthreads();
  int wave = t >> 5, lane = t & 31;
  int row = blockIdx.x * 8 + wave;
  float4 p = make_float4(0.f, 0.f, 0.f, 0.f);
  if (row < N_NODES) p = *(const float4*)(h + (size_t)row * HID + lane * 4);
  atomicAdd(&sacc[lane * 4 + 0], p.x);
  atomicAdd(&sacc[lane * 4 + 1], p.y);
  atomicAdd(&sacc[lane * 4 + 2], p.z);
  atomicAdd(&sacc[lane * 4 + 3], p.w);
  __syncthreads();
  if (t < HID) unsafeAtomicAdd(&acc[t], sacc[t]);
}

__global__ void final_kernel(const float* __restrict__ acc, const float* __restrict__ W_out,
                             const float* __restrict__ b_out, float* __restrict__ out) {
  int lane = threadIdx.x;  // 32
  float p = 0.f;
  for (int i = lane; i < HID; i += 32)
    p += (acc[i] / (float)N_NODES) * W_out[i];
#pragma unroll
  for (int msk = 1; msk < 32; msk <<= 1) p += __shfl_xor(p, msk, 32);
  if (lane == 0) out[0] = (p + b_out[0]) / 50.f;
}

// ---------------------------------------------------------------------------
extern "C" void kernel_launch(void* const* d_in, const int* in_sizes, int n_in,
                              void* d_out, int out_size, void* d_ws, size_t ws_size,
                              hipStream_t stream) {
  const float* x     = (const float*)d_in[0];
  const int*   ei    = (const int*)d_in[1];
  const float* ew    = (const float*)d_in[2];
  const float* ea    = (const float*)d_in[3];
  const float* W_in  = (const float*)d_in[4];
  const float* b_in  = (const float*)d_in[5];
  const float* W_e1  = (const float*)d_in[6];
  const float* b_e1  = (const float*)d_in[7];
  const float* W_e2  = (const float*)d_in[8];
  const float* b_e2  = (const float*)d_in[9];
  const float* W_as  = (const float*)d_in[10];
  const float* b_as  = (const float*)d_in[11];
  const float* W_ad  = (const float*)d_in[12];
  const float* b_ad  = (const float*)d_in[13];
  const float* W_m   = (const float*)d_in[14];
  const float* b_m   = (const float*)d_in[15];
  const float* W_r   = (const float*)d_in[16];
  const float* b_r   = (const float*)d_in[17];
  const float* W_out = (const float*)d_in[18];
  const float* b_out = (const float*)d_in[19];
  float* out = (float*)d_out;

  // Workspace layout (floats). Total ~35.4M floats (~142 MB).
  float* ws = (float*)d_ws;
  size_t off = 0;
  float* sb   = ws + off; off += (size_t)NLAYERS * N_EDGES;  // 9.6M
  float* hA   = ws + off; off += (size_t)N_NODES * HID;
  float* hB   = ws + off; off += (size_t)N_NODES * HID;
  float* msg  = ws + off; off += (size_t)N_NODES * HID;
  float* agg  = ws + off; off += (size_t)N_NODES * HID;
  float* ssrc = ws + off; off += N_NODES;
  float* sdst = ws + off; off += N_NODES;
  float* v_e2 = ws + off; off += NLAYERS * HID;
  float* v_as = ws + off; off += NLAYERS * HID;
  float* v_ad = ws + off; off += NLAYERS * HID;
  float* c_e2 = ws + off; off += NLAYERS;
  float* c_as = ws + off; off += NLAYERS;
  float* c_ad = ws + off; off += NLAYERS;
  float* acc  = ws + off; off += HID;

  // 0) column-sum folding of W_e2 / W_as / W_ad
  prep_kernel<<<NLAYERS, HID, 0, stream>>>(W_e2, b_e2, W_as, b_as, W_ad, b_ad,
                                           v_e2, c_e2, v_as, c_as, v_ad, c_ad);

  // 1) all-layer edge attention bias in one pass over edge_attr (WMMA)
  {
    int waves  = N_EDGES / 16;                    // 100000 tiles
    int blocks = (waves * 32 + 255) / 256;        // 12500
    edge_bias_kernel<<<blocks, 256, 0, stream>>>(ea, W_e1, b_e1, v_e2, c_e2, sb);
  }

  // 2) input projection h0 = x @ W_in^T + b_in  (WMMA, K=64)
  int tiles   = N_NODES / 16;                     // 3125
  int gblocks = (tiles + 7) / 8;                  // 391 blocks x 8 waves
  gemm_kernel<D_INP><<<gblocks, 256, 0, stream>>>(x, nullptr, W_in, b_in, hA, 0, 0);

  float* hcur = hA;
  float* hnext = hB;
  for (int l = 0; l < NLAYERS; ++l) {
    // msg_n = relu(h) @ W_m^T + b_m        (WMMA, K=128)
    gemm_kernel<HID><<<gblocks, 256, 0, stream>>>(
        hcur, nullptr, W_m + (size_t)l * HID * HID, b_m + l * HID, msg, 1, 0);
    // s_src / s_dst matvecs
    scal_kernel<<<(N_NODES + 7) / 8, 256, 0, stream>>>(hcur, v_as, c_as, v_ad, c_ad,
                                                       l, ssrc, sdst);
    // agg = 0
    zero_kernel<<<1024, 256, 0, stream>>>(agg, (size_t)N_NODES * HID);
    // gather * gate * scatter-add
    {
      size_t waves = N_EDGES;
      int blocks = (int)((waves * 32 + 255) / 256);  // 200000
      edge_kernel<<<blocks, 256, 0, stream>>>(ei, ew, sb, ssrc, sdst, msg, agg, l);
    }
    // h = relu((relu(h) + agg) @ W_r^T + b_r)     (WMMA, K=128)
    gemm_kernel<HID><<<gblocks, 256, 0, stream>>>(
        hcur, agg, W_r + (size_t)l * HID * HID, b_r + l * HID, hnext, 1, 1);
    float* tmp = hcur; hcur = hnext; hnext = tmp;
  }

  // 3) graph readout
  zero_kernel<<<1, 128, 0, stream>>>(acc, (size_t)HID);
  colsum_kernel<<<(N_NODES + 7) / 8, 256, 0, stream>>>(hcur, acc);
  final_kernel<<<1, 32, 0, stream>>>(acc, W_out, b_out, out);
}
